// QuadrupletLoss_40269613368004
// MI455X (gfx1250) — compile-verified, matrix operand order checked by reading, not run on hardware
//
#include <hip/hip_runtime.h>
#include <hip/hip_bf16.h>

// ---------------------------------------------------------------------------
// QuadrupletLoss for MI455X (gfx1250, wave32, WMMA)
//
// b=48, s=32, d=512 -> N = 1536 rows. The whole loss reduces to:
//   1) row L2-normalize (1536 x 512)
//   2) Gram G = Xn @ Xn^T  (1536x1536x512 GEMM) via V_WMMA_F32_16X16X4_F32
//   3) per-(i,j) 64x64 distance matrix -> softmax/min/kth-largest epilogue
//   4) hard mining + margin losses -> scalar
//
// Gram kernel: one wave computes a 16(M) x 64(N) strip with 4 accumulators,
// reusing each A fragment across 4 WMMAs (1.25 loads per WMMA).
// ---------------------------------------------------------------------------

#define B48   48
#define S32   32
#define D512  512
#define NROW  (B48 * S32)          // 1536
#define NTILE (NROW / 16)          // 96
#define NSTRP (NTILE / 4)          // 24 strips of 64 columns
#define NUMI  4                    // NUM_INSTANCES
#define EPSF  1e-12f

typedef float     v2f  __attribute__((ext_vector_type(2)));
typedef float     v8f  __attribute__((ext_vector_type(8)));
typedef _Float16  v16h __attribute__((ext_vector_type(16)));

#if defined(__has_builtin)
#  if __has_builtin(__builtin_amdgcn_wmma_f32_16x16x4_f32)
#    define USE_WMMA_F32 1
#  else
#    define USE_WMMA_F32 0
#  endif
#else
#  define USE_WMMA_F32 0
#endif

// ---- wave32 shuffle reductions -------------------------------------------
__device__ __forceinline__ float wave_max32(float v) {
  for (int o = 16; o > 0; o >>= 1) v = fmaxf(v, __shfl_xor(v, o, 32));
  return v;
}
__device__ __forceinline__ float wave_min32(float v) {
  for (int o = 16; o > 0; o >>= 1) v = fminf(v, __shfl_xor(v, o, 32));
  return v;
}
__device__ __forceinline__ float wave_sum32(float v) {
  for (int o = 16; o > 0; o >>= 1) v += __shfl_xor(v, o, 32);
  return v;
}

// ---------------------------------------------------------------------------
// Kernel 1: row L2-normalize. One wave per row (8 waves / 256-thread block).
// Writes Xn (f32) and Xh (f16 copy, used only by the fallback WMMA path).
// ---------------------------------------------------------------------------
__global__ void ql_normalize(const float* __restrict__ X,
                             float* __restrict__ Xn,
                             _Float16* __restrict__ Xh) {
  const int lane = threadIdx.x & 31;
  const int wid  = threadIdx.x >> 5;
  const int row  = blockIdx.x * 8 + wid;
  if (row >= NROW) return;

  const float* src = X + (size_t)row * D512 + lane * 16;
  float v[16];
  #pragma unroll
  for (int q = 0; q < 4; ++q) {
    const float4 t = ((const float4*)src)[q];
    v[4*q+0] = t.x; v[4*q+1] = t.y; v[4*q+2] = t.z; v[4*q+3] = t.w;
  }

  float ss = 0.f;
  #pragma unroll
  for (int q = 0; q < 16; ++q) ss += v[q] * v[q];
  ss = wave_sum32(ss);
  const float inv = 1.0f / fmaxf(sqrtf(ss), EPSF);
  #pragma unroll
  for (int q = 0; q < 16; ++q) v[q] *= inv;

  float* dst = Xn + (size_t)row * D512 + lane * 16;
  #pragma unroll
  for (int q = 0; q < 4; ++q) {
    float4 t; t.x = v[4*q+0]; t.y = v[4*q+1]; t.z = v[4*q+2]; t.w = v[4*q+3];
    ((float4*)dst)[q] = t;
  }
  _Float16* dh = Xh + (size_t)row * D512 + lane * 16;
  #pragma unroll
  for (int q = 0; q < 16; ++q) dh[q] = (_Float16)v[q];
}

// ---------------------------------------------------------------------------
// Kernel 2: Gram G = Xn * Xn^T (1536 x 1536, K = 512) via WMMA.
// One wave computes a 16x64 strip (4 accumulators); 8 waves per block.
//
// 32-bit A fragment (ISA 7.12.2): lanes 0-15 hold {K=k,k+1} for M=lane,
// lanes 16-31 hold {K=k+2,k+3} for M=lane-16. B = Xn^T, so the B fragment
// of column tile tn is the identical pattern read from rows tn*16.. of Xn;
// the 4 B tiles of a strip sit at fixed 16*512*4 = 32768-byte offsets.
// ---------------------------------------------------------------------------
__global__ void ql_gram(const float* __restrict__ Xn,
                        const _Float16* __restrict__ Xh,
                        float* __restrict__ G) {
  const int lane = threadIdx.x & 31;
  const int wid  = threadIdx.x >> 5;
  const int strip = blockIdx.x * 8 + wid;        // 96 * 24 strips
  if (strip >= NTILE * NSTRP) return;
  const int tm  = strip / NSTRP;                 // 16-row block of A
  const int tnb = strip % NSTRP;                 // 64-col block of G

  const int half = lane >> 4;                    // 0: lanes 0-15, 1: 16-31
  const int l    = lane & 15;

  v8f acc0 = {0.f,0.f,0.f,0.f,0.f,0.f,0.f,0.f};
  v8f acc1 = acc0, acc2 = acc0, acc3 = acc0;

#if USE_WMMA_F32
  const float* Arow = Xn + (size_t)(tm  * 16 + l) * D512;
  const float* Brow = Xn + (size_t)(tnb * 64 + l) * D512;   // + q*16*512
  #pragma unroll 2
  for (int k = 0; k < D512; k += 4) {
    const int kk = k + 2 * half;                 // even -> 8B aligned
    const float2 af = *(const float2*)(Arow + kk);
    const float2 b0 = *(const float2*)(Brow + kk);
    const float2 b1 = *(const float2*)(Brow + kk + 16 * D512);
    const float2 b2 = *(const float2*)(Brow + kk + 32 * D512);
    const float2 b3 = *(const float2*)(Brow + kk + 48 * D512);
    v2f a;  a.x  = af.x; a.y  = af.y;
    v2f f0; f0.x = b0.x; f0.y = b0.y;
    v2f f1; f1.x = b1.x; f1.y = b1.y;
    v2f f2; f2.x = b2.x; f2.y = b2.y;
    v2f f3; f3.x = b3.x; f3.y = b3.y;
    acc0 = __builtin_amdgcn_wmma_f32_16x16x4_f32(false, a, false, f0,
                                                 (short)0, acc0, false, false);
    acc1 = __builtin_amdgcn_wmma_f32_16x16x4_f32(false, a, false, f1,
                                                 (short)0, acc1, false, false);
    acc2 = __builtin_amdgcn_wmma_f32_16x16x4_f32(false, a, false, f2,
                                                 (short)0, acc2, false, false);
    acc3 = __builtin_amdgcn_wmma_f32_16x16x4_f32(false, a, false, f3,
                                                 (short)0, acc3, false, false);
  }
  (void)Xh;
#else
  // Fallback: f16 inputs, f32 accumulate, K-step 32.
  // 16-bit A layout: lanes 0-15 hold K = {k..k+7, k+16..k+23},
  // lanes 16-31 hold K = {k+8..k+15, k+24..k+31}.
  const _Float16* Arow = Xh + (size_t)(tm  * 16 + l) * D512;
  const _Float16* Brow = Xh + (size_t)(tnb * 64 + l) * D512;
  for (int k = 0; k < D512; k += 32) {
    const int kb = k + 8 * half;
    v16h a, b[4];
    #pragma unroll
    for (int q = 0; q < 8; ++q) {
      a[q]     = Arow[kb + q];
      a[8 + q] = Arow[kb + 16 + q];
      #pragma unroll
      for (int s = 0; s < 4; ++s) {
        b[s][q]     = Brow[(size_t)s * 16 * D512 + kb + q];
        b[s][8 + q] = Brow[(size_t)s * 16 * D512 + kb + 16 + q];
      }
    }
    acc0 = __builtin_amdgcn_wmma_f32_16x16x32_f16(false, a, false, b[0],
                                                  (short)0, acc0, false, false);
    acc1 = __builtin_amdgcn_wmma_f32_16x16x32_f16(false, a, false, b[1],
                                                  (short)0, acc1, false, false);
    acc2 = __builtin_amdgcn_wmma_f32_16x16x32_f16(false, a, false, b[2],
                                                  (short)0, acc2, false, false);
    acc3 = __builtin_amdgcn_wmma_f32_16x16x32_f16(false, a, false, b[3],
                                                  (short)0, acc3, false, false);
  }
  (void)Xn;
#endif

  // C/D layout: VGPR r, lanes 0-15 -> (M = r, N = lane),
  //             lanes 16-31 -> (M = r+8, N = lane-16).
  const int gm0 = tm * 16 + half * 8;
  const int gn0 = tnb * 64 + l;
  #pragma unroll
  for (int r = 0; r < 8; ++r) {
    float* gp = G + (size_t)(gm0 + r) * NROW + gn0;
    gp[0]  = acc0[r];
    gp[16] = acc1[r];
    gp[32] = acc2[r];
    gp[48] = acc3[r];
  }
}

// ---------------------------------------------------------------------------
// Kernel 3: per-pair value. One block (256 thr = 8 waves) per (i,j).
// D[a][c] = sqrt(max(2 - 2*G[ga][gc], EPS)) for the 64x64 concat matrix,
// staged in LDS. Row/col softmax + cross-block min + kth-largest selection.
// ---------------------------------------------------------------------------
__global__ void ql_pair(const float* __restrict__ G,
                        float* __restrict__ dist) {
  const int i = blockIdx.x / B48;
  const int j = blockIdx.x % B48;
  if (i == j) {                       // uniform across block: safe early-out
    if (threadIdx.x == 0) dist[i * B48 + j] = 0.0f;
    return;
  }

  __shared__ float Dsh[64 * 64];
  __shared__ float rmin_sh[32];
  __shared__ float cmin_sh[32];

  const int tid  = threadIdx.x;
  const int lane = tid & 31;
  const int w    = tid >> 5;

  // stage D
  for (int idx = tid; idx < 64 * 64; idx += 256) {
    const int a = idx >> 6, c = idx & 63;
    const int ga = (a < 32) ? (i * S32 + a) : (j * S32 + (a - 32));
    const int gc = (c < 32) ? (i * S32 + c) : (j * S32 + (c - 32));
    const float g = G[(size_t)ga * NROW + gc];
    Dsh[idx] = sqrtf(fmaxf(2.0f - 2.0f * g, EPSF));
  }
  __syncthreads();

  // row softmax over full 64, min over cross-block cols [32,64)
  for (int rr = w; rr < 32; rr += 8) {
    const float d0 = Dsh[rr * 64 + lane];
    const float d1 = Dsh[rr * 64 + 32 + lane];
    const float mx = wave_max32(fmaxf(d0, d1));
    const float sm = wave_sum32(expf(d0 - mx) + expf(d1 - mx));
    const float rv = expf(d1 - mx) / sm;          // col = 32 + lane
    const float mn = wave_min32(rv);
    if (lane == 0) rmin_sh[rr] = mn;
  }
  // col softmax over full 64, min over rows [0,32)
  for (int cc = w; cc < 32; cc += 8) {
    const int c = 32 + cc;
    const float d0 = Dsh[lane * 64 + c];          // rows 0..31
    const float d1 = Dsh[(32 + lane) * 64 + c];   // rows 32..63
    const float mx = wave_max32(fmaxf(d0, d1));
    const float sm = wave_sum32(expf(d0 - mx) + expf(d1 - mx));
    const float cv = expf(d0 - mx) / sm;          // row = lane (0..31)
    const float mn = wave_min32(cv);
    if (lane == 0) cmin_sh[cc] = mn;
  }
  __syncthreads();

  if (w == 0) {
    const float NEG = -__builtin_inff();
    float vr = rmin_sh[lane], vc = cmin_sh[lane];
    float r3 = 0.f, r6 = 0.f, c3 = 0.f, c6 = 0.f;
    #pragma unroll
    for (int it = 1; it <= 6; ++it) {
      float mr = wave_max32(vr);
      float mc = wave_max32(vc);
      if (it == 3) { r3 = mr; c3 = mc; }
      if (it == 6) { r6 = mr; c6 = mc; }
      unsigned long long br = __ballot(vr == mr);
      unsigned long long bc = __ballot(vc == mc);
      if (lane == __ffsll(br) - 1) vr = NEG;
      if (lane == __ffsll(bc) - 1) vc = NEG;
    }
    const bool same = (i / NUMI) == (j / NUMI);
    const float val = same ? fmaxf(r6, c6) : fminf(r3, c3);
    if (lane == 0) dist[i * B48 + j] = val;
  }
}

// ---------------------------------------------------------------------------
// Kernel 4: hard mining + margin losses -> scalar. Single 64-thread block.
// ---------------------------------------------------------------------------
__global__ void ql_loss(const float* __restrict__ dist,
                        float* __restrict__ out) {
  __shared__ float ap_s[B48];
  __shared__ float an_s[B48];
  __shared__ int   ix_s[B48];
  __shared__ float sum_s[64];

  const int t = threadIdx.x;
  if (t < B48) {
    float ap = -__builtin_inff();
    float an =  __builtin_inff();
    int   ix = 0;
    for (int c = 0; c < B48; ++c) {
      const bool  m = (t / NUMI) == (c / NUMI);
      const float v = dist[t * B48 + c];
      if (m) {
        ap = fmaxf(ap, v);
      } else if (v < an) {            // strict '<' => first-occurrence argmin
        an = v; ix = c;
      }
    }
    ap_s[t] = ap; an_s[t] = an; ix_s[t] = ix;
  }
  __syncthreads();

  float contrib = 0.0f;
  if (t < B48) {
    const float ann = an_s[ix_s[t]];
    const float l1 = fmaxf(0.0f, 1.0f + ap_s[t] - an_s[t]);   // MARGIN1
    const float l2 = fmaxf(0.0f, 0.5f + ap_s[t] - ann);       // MARGIN2
    contrib = l1 + l2;                 // ALPHA = BETA = 1
  }
  sum_s[t] = contrib;
  __syncthreads();

  if (t == 0) {
    float s = 0.0f;
    for (int k = 0; k < B48; ++k) s += sum_s[k];
    out[0] = s / (float)B48;           // mean(l1) + mean(l2)
  }
}

// ---------------------------------------------------------------------------
// Launch
// ---------------------------------------------------------------------------
extern "C" void kernel_launch(void* const* d_in, const int* in_sizes, int n_in,
                              void* d_out, int out_size, void* d_ws, size_t ws_size,
                              hipStream_t stream) {
  (void)in_sizes; (void)n_in; (void)out_size; (void)ws_size;

  const float* X   = (const float*)d_in[0];
  float*       out = (float*)d_out;

  char* w = (char*)d_ws;
  float*    Xn   = (float*)(w + 0);
  _Float16* Xh   = (_Float16*)(w + (size_t)NROW * D512 * sizeof(float));
  float*    G    = (float*)(w + (size_t)NROW * D512 * (sizeof(float) + sizeof(_Float16)));
  float*    dist = (float*)((char*)G + (size_t)NROW * NROW * sizeof(float));

  ql_normalize<<<NROW / 8, 256, 0, stream>>>(X, Xn, Xh);
  ql_gram<<<(NTILE * NSTRP) / 8, 256, 0, stream>>>(Xn, Xh, G);
  ql_pair<<<B48 * B48, 256, 0, stream>>>(G, dist);
  ql_loss<<<1, 64, 0, stream>>>(dist, out);
}
